// HunYuanMoEV1Moe_37331855736952
// MI455X (gfx1250) — matrix operand order repaired and checked by reading, not compiled, not run on hardware
//
#include <hip/hip_runtime.h>

// ---------------- problem constants (match reference) ----------------
#define NE  8      // experts
#define HID 2048   // hidden dim (D)
#define FF  4096   // ffn dim (F)
#define TT  2048   // tokens = B*S

// ---------------- WMMA types ----------------
typedef __bf16 v16bf __attribute__((ext_vector_type(16)));
typedef __bf16 v4bf  __attribute__((ext_vector_type(4)));
typedef float  v4f   __attribute__((ext_vector_type(4)));
typedef float  v8f   __attribute__((ext_vector_type(8)));

struct Q2 { uint4 a, b; };

// Build a 16x bf16 WMMA fragment register set from two 16-byte LDS chunks.
static __device__ __forceinline__ v16bf frag32(const unsigned short* p0,
                                               const unsigned short* p1) {
  Q2 q;
  q.a = *(const uint4*)p0;
  q.b = *(const uint4*)p1;
  return __builtin_bit_cast(v16bf, q);
}

// fp32 -> bf16 via native conversion (RNE), single element
static __device__ __forceinline__ unsigned short f2bf(float f) {
  __bf16 b = (__bf16)f;
  return __builtin_bit_cast(unsigned short, b);
}

// fp32x4 -> bf16x4 via native packed conversion (fptrunc <4 x float> -> <4 x bfloat>)
static __device__ __forceinline__ uint2 pack4(float4 f) {
  v4f x = { f.x, f.y, f.z, f.w };
  v4bf b = __builtin_convertvector(x, v4bf);
  return __builtin_bit_cast(uint2, b);
}

// ---------------- x fp32 -> bf16 ----------------
__global__ __launch_bounds__(256) void k_convert(const float* __restrict__ x,
                                                 unsigned short* __restrict__ xbf) {
  int i = (blockIdx.x * 256 + threadIdx.x) * 8;
  float4 a = *(const float4*)(x + i);
  float4 b = *(const float4*)(x + i + 4);
  uint2 pa = pack4(a), pb = pack4(b);
  *(uint4*)(xbf + i) = make_uint4(pa.x, pa.y, pb.x, pb.y);
}

// ---------------- router: logits, softmax, top-2, combine weights ----------------
__global__ __launch_bounds__(256) void k_router(const float* __restrict__ x,
                                                const float* __restrict__ wg,
                                                float* __restrict__ comb) {
  const int lane = threadIdx.x & 31;
  const int wid  = threadIdx.x >> 5;
  const int t    = blockIdx.x * 8 + wid;   // one wave32 per token
  const float* xr = x + (size_t)t * HID;

  float acc[NE];
#pragma unroll
  for (int e = 0; e < NE; ++e) acc[e] = 0.f;
  for (int k = lane; k < HID; k += 32) {
    float xv = xr[k];
#pragma unroll
    for (int e = 0; e < NE; ++e) acc[e] += xv * wg[e * HID + k];
  }
#pragma unroll
  for (int e = 0; e < NE; ++e)
#pragma unroll
    for (int off = 16; off > 0; off >>= 1)
      acc[e] += __shfl_xor(acc[e], off, 32);

  if (lane == 0) {
    float mx = acc[0];
#pragma unroll
    for (int e = 1; e < NE; ++e) mx = fmaxf(mx, acc[e]);
    float p[NE];
#pragma unroll
    for (int e = 0; e < NE; ++e) p[e] = __expf(acc[e] - mx);
    int i1 = 0;
#pragma unroll
    for (int e = 1; e < NE; ++e) if (p[e] > p[i1]) i1 = e;
    int i2 = (i1 == 0) ? 1 : 0;
#pragma unroll
    for (int e = 0; e < NE; ++e) if (e != i1 && p[e] > p[i2]) i2 = e;
    float s = p[i1] + p[i2];
#pragma unroll
    for (int e = 0; e < NE; ++e) comb[t * NE + e] = 0.f;
    comb[t * NE + i1] = p[i1] / s;
    comb[t * NE + i2] = p[i2] / s;
  }
}

// ---------------- dispatch build ----------------
__global__ void k_zero_meta(int* counts, int* cursors) {
  if (threadIdx.x < NE) { counts[threadIdx.x] = 0; cursors[threadIdx.x] = 0; }
}
__global__ __launch_bounds__(256) void k_count(const float* __restrict__ comb,
                                               int* __restrict__ counts) {
  int t = blockIdx.x * 256 + threadIdx.x;
  if (t < TT)
    for (int e = 0; e < NE; ++e)
      if (comb[t * NE + e] > 0.f) atomicAdd(&counts[e], 1);
}
__global__ void k_scan(const int* __restrict__ counts, int* offsets, int* cursors) {
  if (threadIdx.x == 0) {
    int s = 0;
    for (int e = 0; e < NE; ++e) { offsets[e] = s; cursors[e] = s; s += counts[e]; }
  }
}
__global__ __launch_bounds__(256) void k_fill(const float* __restrict__ comb,
                                              int* __restrict__ cursors,
                                              int* __restrict__ toklist) {
  int t = blockIdx.x * 256 + threadIdx.x;
  if (t < TT)
    for (int e = 0; e < NE; ++e)
      if (comb[t * NE + e] > 0.f) {
        int p = atomicAdd(&cursors[e], 1);
        toklist[p] = t;
      }
}

__global__ __launch_bounds__(256) void k_zero_out(float* __restrict__ out) {
  int i = blockIdx.x * 256 + threadIdx.x;
  float4 z = {0.f, 0.f, 0.f, 0.f};
  *(float4*)(out + (size_t)i * 4) = z;
}

// ---------------- fused gate+up GEMM: h = silu(x Wg^T) * (x Wu^T) ----------------
// block: 128 threads (4 waves), tile 64(M) x 64(N), K-step 32, bf16 WMMA,
// 1-deep software pipeline (next K-tile staged into regs during WMMA work).
__global__ __launch_bounds__(128) void k_gateup(
    const unsigned short* __restrict__ xbf,
    const float* __restrict__ WgA, const float* __restrict__ WuA,
    unsigned short* __restrict__ hbuf,
    const int* __restrict__ toklist, const int* __restrict__ counts,
    const int* __restrict__ offsets, long wstride) {
  __shared__ __align__(16) unsigned short lA[64 * 32];
  __shared__ __align__(16) unsigned short lG[64 * 32];
  __shared__ __align__(16) unsigned short lU[64 * 32];

  const int e       = blockIdx.z;
  const int count   = counts ? counts[e] : TT;
  const int rowTile = blockIdx.x * 64;           // x = rows: consecutive blocks share weights
  if (rowTile >= count) return;
  const int segOff  = offsets ? offsets[e] : 0;
  const float* Wg   = WgA + (size_t)e * (size_t)wstride;
  const float* Wu   = WuA + (size_t)e * (size_t)wstride;
  const int nTile   = blockIdx.y * 64;

  const int tid  = (int)threadIdx.x;
  const int lane = tid & 31;
  const int wid  = tid >> 5;
  const int half = lane >> 4;
  const int l16  = lane & 15;

  // per-thread staging coordinates
  const int ar0 = tid >> 2, ak0 = (tid & 3) * 8;        // A chunk 0
  const int ar1 = (tid + 128) >> 2, ak1 = ak0;          // A chunk 1
  const int wn[4] = { tid >> 3, (tid + 128) >> 3, (tid + 256) >> 3, (tid + 384) >> 3 };
  const int wk = (tid & 7) * 4;

  uint4  aR[2];
  float4 gR[4], uR[4];

  auto stageLoad = [&](int kt) {
    // A: 64x32 bf16 (gathered rows, zero-pad past count)
    {
      uint4 v0 = {0u,0u,0u,0u}, v1 = {0u,0u,0u,0u};
      int g0 = rowTile + ar0, g1 = rowTile + ar1;
      if (g0 < count) {
        int tok = toklist ? toklist[segOff + g0] : g0;
        v0 = *(const uint4*)(xbf + (size_t)tok * HID + kt + ak0);
      }
      if (g1 < count) {
        int tok = toklist ? toklist[segOff + g1] : g1;
        v1 = *(const uint4*)(xbf + (size_t)tok * HID + kt + ak1);
      }
      aR[0] = v0; aR[1] = v1;
    }
    // weights: 64x32 fp32 each
#pragma unroll
    for (int i = 0; i < 4; ++i) {
      size_t woff = (size_t)(nTile + wn[i]) * HID + kt + wk;
      gR[i] = *(const float4*)(Wg + woff);
      uR[i] = *(const float4*)(Wu + woff);
      if (kt + 64 <= HID) {                 // keep the weight stream ahead
        __builtin_prefetch(Wg + woff + 32, 0, 1);
        __builtin_prefetch(Wu + woff + 32, 0, 1);
      }
    }
  };
  auto stageStore = [&]() {
    *(uint4*)(lA + ar0 * 32 + ak0) = aR[0];
    *(uint4*)(lA + ar1 * 32 + ak1) = aR[1];
#pragma unroll
    for (int i = 0; i < 4; ++i) {
      *(uint2*)(lG + wn[i] * 32 + wk) = pack4(gR[i]);
      *(uint2*)(lU + wn[i] * 32 + wk) = pack4(uR[i]);
    }
  };

  v8f accG[4] = {}, accU[4] = {};

  stageLoad(0);
  stageStore();
  for (int kt = 0; kt < HID; kt += 32) {
    __syncthreads();                       // LDS tile ready
    if (kt + 32 < HID) stageLoad(kt + 32); // overlap next-tile loads with WMMA

    const unsigned short* aRow = lA + (wid * 16 + l16) * 32;
    v16bf af = frag32(aRow + half * 16, aRow + half * 16 + 8);
    v16bf bg[4], bu[4];
#pragma unroll
    for (int j = 0; j < 4; ++j) {          // all fragment loads first: one wait, no per-WMMA stalls
      const unsigned short* pG = lG + (j * 16 + l16) * 32 + half * 16;
      const unsigned short* pU = lU + (j * 16 + l16) * 32 + half * 16;
      bg[j] = frag32(pG, pG + 8);
      bu[j] = frag32(pU, pU + 8);
    }
#pragma unroll
    for (int j = 0; j < 4; ++j) {
      accG[j] = __builtin_amdgcn_wmma_f32_16x16x32_bf16(false, af, false, bg[j],
                                                        (short)0, accG[j], false, false);
      accU[j] = __builtin_amdgcn_wmma_f32_16x16x32_bf16(false, af, false, bu[j],
                                                        (short)0, accU[j], false, false);
    }
    __syncthreads();                       // all waves done reading LDS
    if (kt + 32 < HID) stageStore();
  }

  // epilogue: silu(gate) * up -> bf16 h
  const int strip = rowTile + wid * 16;
#pragma unroll
  for (int j = 0; j < 4; ++j)
#pragma unroll
    for (int r = 0; r < 8; ++r) {
      int m = half * 8 + r;
      int grow = strip + m;
      if (grow < count) {
        float g = accG[j][r], u = accU[j][r];
        float h = (g / (1.0f + __expf(-g))) * u;
        hbuf[(size_t)(segOff + grow) * FF + nTile + j * 16 + l16] = f2bf(h);
      }
    }
}

// ---------------- down GEMM: out[tok] += scale * (h Wd^T) ----------------
__global__ __launch_bounds__(128) void k_down(
    const unsigned short* __restrict__ hbuf,
    const float* __restrict__ WdA,
    float* __restrict__ out,
    const int* __restrict__ toklist, const int* __restrict__ counts,
    const int* __restrict__ offsets, const float* __restrict__ comb,
    long wstride) {
  __shared__ __align__(16) unsigned short lA[64 * 32];
  __shared__ __align__(16) unsigned short lB[64 * 32];

  const int e       = blockIdx.z;
  const int count   = counts ? counts[e] : TT;
  const int rowTile = blockIdx.x * 64;
  if (rowTile >= count) return;
  const int segOff  = offsets ? offsets[e] : 0;
  const float* Wd   = WdA + (size_t)e * (size_t)wstride;
  const int nTile   = blockIdx.y * 64;

  const int tid  = (int)threadIdx.x;
  const int lane = tid & 31;
  const int wid  = tid >> 5;
  const int half = lane >> 4;
  const int l16  = lane & 15;

  const int ar0 = tid >> 2, ak0 = (tid & 3) * 8;
  const int ar1 = (tid + 128) >> 2, ak1 = ak0;
  const int wn[4] = { tid >> 3, (tid + 128) >> 3, (tid + 256) >> 3, (tid + 384) >> 3 };
  const int wk = (tid & 7) * 4;

  uint4  aR[2];
  float4 bR[4];

  auto stageLoad = [&](int kt) {
    uint4 v0 = {0u,0u,0u,0u}, v1 = {0u,0u,0u,0u};
    int g0 = rowTile + ar0, g1 = rowTile + ar1;
    if (g0 < count) v0 = *(const uint4*)(hbuf + (size_t)(segOff + g0) * FF + kt + ak0);
    if (g1 < count) v1 = *(const uint4*)(hbuf + (size_t)(segOff + g1) * FF + kt + ak1);
    aR[0] = v0; aR[1] = v1;
#pragma unroll
    for (int i = 0; i < 4; ++i) {
      size_t woff = (size_t)(nTile + wn[i]) * FF + kt + wk;
      bR[i] = *(const float4*)(Wd + woff);
      if (kt + 64 <= FF) __builtin_prefetch(Wd + woff + 32, 0, 1);
    }
  };
  auto stageStore = [&]() {
    *(uint4*)(lA + ar0 * 32 + ak0) = aR[0];
    *(uint4*)(lA + ar1 * 32 + ak1) = aR[1];
#pragma unroll
    for (int i = 0; i < 4; ++i)
      *(uint2*)(lB + wn[i] * 32 + wk) = pack4(bR[i]);
  };

  v8f acc[4] = {};

  stageLoad(0);
  stageStore();
  for (int kt = 0; kt < FF; kt += 32) {
    __syncthreads();
    if (kt + 32 < FF) stageLoad(kt + 32);

    const unsigned short* aRow = lA + (wid * 16 + l16) * 32;
    v16bf af = frag32(aRow + half * 16, aRow + half * 16 + 8);
    v16bf bv[4];
#pragma unroll
    for (int j = 0; j < 4; ++j) {
      const unsigned short* pB = lB + (j * 16 + l16) * 32 + half * 16;
      bv[j] = frag32(pB, pB + 8);
    }
#pragma unroll
    for (int j = 0; j < 4; ++j)
      acc[j] = __builtin_amdgcn_wmma_f32_16x16x32_bf16(false, af, false, bv[j],
                                                       (short)0, acc[j], false, false);
    __syncthreads();
    if (kt + 32 < FF) stageStore();
  }

  const int strip = rowTile + wid * 16;
#pragma unroll
  for (int j = 0; j < 4; ++j)
#pragma unroll
    for (int r = 0; r < 8; ++r) {
      int m = half * 8 + r;
      int grow = strip + m;
      if (grow < count) {
        int tok = toklist ? toklist[segOff + grow] : grow;
        float s = comb ? comb[tok * NE + e] : 1.0f;
        atomicAdd(out + (size_t)tok * HID + nTile + j * 16 + l16, acc[j][r] * s);
      }
    }
}

// ---------------- launcher ----------------
extern "C" void kernel_launch(void* const* d_in, const int* in_sizes, int n_in,
                              void* d_out, int out_size, void* d_ws, size_t ws_size,
                              hipStream_t stream) {
  const float* x       = (const float*)d_in[0];
  const float* wg      = (const float*)d_in[1];
  const float* sh_gate = (const float*)d_in[2];
  const float* sh_up   = (const float*)d_in[3];
  const float* sh_down = (const float*)d_in[4];
  const float* ex_gate = (const float*)d_in[5];
  const float* ex_up   = (const float*)d_in[6];
  const float* ex_down = (const float*)d_in[7];
  float* out = (float*)d_out;
  (void)in_sizes; (void)n_in; (void)out_size; (void)ws_size;

  char* ws = (char*)d_ws;
  size_t off = 0;
  auto alloc = [&](size_t bytes) -> void* {
    off = (off + 255) & ~(size_t)255;
    void* p = ws + off;
    off += bytes;
    return p;
  };
  unsigned short* xbf  = (unsigned short*)alloc((size_t)TT * HID * 2);   //  8 MB
  float* comb          = (float*)alloc((size_t)TT * NE * 4);             // 64 KB
  int* counts          = (int*)alloc(NE * 4);
  int* offsets         = (int*)alloc(NE * 4);
  int* cursors         = (int*)alloc(NE * 4);
  int* toklist         = (int*)alloc((size_t)TT * 2 * 4);                // 16 KB
  unsigned short* h_sh = (unsigned short*)alloc((size_t)TT * FF * 2);    // 16 MB
  unsigned short* h_ex = (unsigned short*)alloc((size_t)TT * 2 * FF * 2);// 32 MB

  k_convert<<<TT * HID / (256 * 8), 256, 0, stream>>>(x, xbf);
  k_router<<<TT / 8, 256, 0, stream>>>(x, wg, comb);
  k_zero_meta<<<1, 32, 0, stream>>>(counts, cursors);
  k_count<<<TT / 256, 256, 0, stream>>>(comb, counts);
  k_scan<<<1, 32, 0, stream>>>(counts, offsets, cursors);
  k_fill<<<TT / 256, 256, 0, stream>>>(comb, cursors, toklist);
  k_zero_out<<<TT * HID / 1024, 256, 0, stream>>>(out);

  // shared branch (all tokens, scale 1)
  k_gateup<<<dim3(TT / 64, FF / 64, 1), 128, 0, stream>>>(
      xbf, sh_gate, sh_up, h_sh, nullptr, nullptr, nullptr, 0L);
  k_down<<<dim3(TT / 64, HID / 64, 1), 128, 0, stream>>>(
      h_sh, sh_down, out, nullptr, nullptr, nullptr, nullptr, 0L);

  // expert branches (gathered tokens, combine-weight scale)
  k_gateup<<<dim3(TT / 64, FF / 64, NE), 128, 0, stream>>>(
      xbf, ex_gate, ex_up, h_ex, toklist, counts, offsets, (long)FF * HID);
  k_down<<<dim3(TT / 64, HID / 64, NE), 128, 0, stream>>>(
      h_ex, ex_down, out, toklist, counts, offsets, comb, (long)HID * FF);
}
